// RWKVAttention_66013647339848
// MI455X (gfx1250) — matrix operand order repaired and checked by reading, not compiled, not run on hardware
//
#include <hip/hip_runtime.h>
#include <stdint.h>

// ---------------------------------------------------------------------------
// RWKV attention block for MI455X (gfx1250, wave32).
// Heavy GEMMs: v_wmma_f32_16x16x32_bf16, B-tiles streamed to LDS with
// GLOBAL_LOAD_ASYNC_TO_LDS (ASYNCcnt, double-buffered).
// ---------------------------------------------------------------------------

typedef __bf16 bf16;
typedef __attribute__((ext_vector_type(16))) __bf16 v16bf;
typedef __attribute__((ext_vector_type(8)))  float  v8f;

#define HID   2048
#define HEADD 64
#define NHEAD 32
#define BB    4
#define TT    2048
#define MTOT  (BB * TT)       // 8192 rows in the (B*T, HID) matrices

union FragB16 {
    v16bf v;
    uint4 q[2];
};

// A-matrix 16x32 bf16 fragment (row-major source, row stride = `stride` elems).
// Layout: lanes 0-15 -> rows M=0..15, K 0..15 (contiguous)
//         lanes 16-31 -> same rows,  K 16..31 (matches B fragment order).
__device__ __forceinline__ void load_fragA(FragB16& f, const bf16* __restrict__ base,
                                           int stride, int row0, int k0, int lane) {
    const int r = row0 + (lane & 15);
    const int h = lane >> 4;
    const bf16* p = base + (size_t)r * stride + k0 + h * 16;
    f.q[0] = *(const uint4*)(p);
    f.q[1] = *(const uint4*)(p + 8);
}

// B-matrix 32x16 bf16 fragment from an LDS tile stored as [col][k0..31].
// Layout used here (same k order as A): lanes 0-15 -> col N holding K 0..15,
// lanes 16-31 -> same cols holding K 16..31 (contiguous 16 elements).
__device__ __forceinline__ void load_fragB(FragB16& f, const bf16* __restrict__ lds,
                                           int col, int lane) {
    const int h = lane >> 4;
    const bf16* p = lds + col * 32 + h * 16;
    f.q[0] = *(const uint4*)(p);
    f.q[1] = *(const uint4*)(p + 8);
}

// ---------------------------------------------------------------------------
// Generic bf16 GEMM:  C[M x N] = epilogue( A[M x K] * Wt[N x K]^T )
//   A  : bf16 row-major, row stride lda
//   Wt : bf16, pre-transposed weights, row-major [N][K]
// Block = 256 threads (8 waves); wave tile = 16 rows x (NT*16) cols;
// block tile = 128 rows x (NT*16) cols.
// B tile (NT*16 cols x 32 k) streamed memory->LDS via async-load, double
// buffered: s_wait_asynccnt -> barrier -> issue next tile -> WMMA current.
// MODE 0: store bf16 | 1: tanh->bf16 | 2: silu->bf16
// MODE 3: bf16( x + dx*(maa + acc) )   (RWKV token-mix fusion)
// MODE 4: store f32
// ---------------------------------------------------------------------------
template <int MODE, int NT>
__global__ __launch_bounds__(256) void gemm_bf16(
    const bf16* __restrict__ A, int lda,
    const bf16* __restrict__ Wt,
    int N, int K,
    bf16* __restrict__ outB, int ldo,
    float* __restrict__ outF,
    const float* __restrict__ xres,
    const float* __restrict__ dxres,
    const float* __restrict__ maa)
{
    constexpr int TILE = NT * 16 * 32;   // bf16 elems per K-step B tile
    constexpr int EPT  = TILE / 256;     // elems per thread: 4 (NT=2) / 8 (NT=4)
    __shared__ __align__(16) bf16 bt[2][TILE];

    const int lane = threadIdx.x & 31;
    const int wave = threadIdx.x >> 5;
    const int n0   = blockIdx.x * (NT * 16);
    const int row0 = blockIdx.y * 128 + wave * 16;

    // staging coordinates for this thread (contiguous EPT bf16 along K)
    const int scol = (threadIdx.x * EPT) >> 5;
    const int sk   = (threadIdx.x * EPT) & 31;

    auto issue = [&](int kk, int bufsel) {
        const bf16* src = Wt + (size_t)(n0 + scol) * K + kk + sk;
        const unsigned lds =
            (unsigned)(unsigned long long)(void*)&bt[bufsel][scol * 32 + sk];
        const unsigned long long ga = (unsigned long long)src;
        if (EPT == 8)
            asm volatile("global_load_async_to_lds_b128 %0, %1, off"
                         :: "v"(lds), "v"(ga) : "memory");
        else
            asm volatile("global_load_async_to_lds_b64 %0, %1, off"
                         :: "v"(lds), "v"(ga) : "memory");
    };

    v8f acc[NT] = {};
    issue(0, 0);
    int buf = 0;

    for (int kk = 0; kk < K; kk += 32) {
        asm volatile("s_wait_asynccnt 0x0" ::: "memory");
        __syncthreads();                       // all waves' tile parts visible
        if (kk + 32 < K) issue(kk + 32, buf ^ 1);
        if (kk + 32 < K)
            __builtin_prefetch(A + (size_t)(row0 + (lane & 15)) * lda + kk + 32, 0, 3);

        FragB16 a;
        load_fragA(a, A, lda, row0, kk, lane);
#pragma unroll
        for (int c = 0; c < NT; ++c) {
            FragB16 b;
            load_fragB(b, bt[buf], c * 16 + (lane & 15), lane);
            acc[c] = __builtin_amdgcn_wmma_f32_16x16x32_bf16(
                false, a.v, false, b.v, (short)0, acc[c], false, false);
        }
        buf ^= 1;
    }

    const int half = lane >> 4;
    const int cl   = lane & 15;
#pragma unroll
    for (int c = 0; c < NT; ++c) {
        const int cc = n0 + c * 16 + cl;
#pragma unroll
        for (int i = 0; i < 8; ++i) {
            const int r = row0 + i + 8 * half;   // C layout: VGPR i -> M=i (+8 hi lanes)
            float v0 = acc[c][i];
            if (MODE == 1) v0 = tanhf(v0);
            if (MODE == 2) v0 = v0 / (1.f + __expf(-v0));
            if (MODE == 3) {
                const size_t g0 = (size_t)r * HID + cc;
                outB[g0] = (bf16)(xres[g0] + dxres[g0] * (maa[cc] + v0));
            } else if (MODE == 4) {
                outF[(size_t)r * ldo + cc] = v0;
            } else {
                outB[(size_t)r * ldo + cc] = (bf16)v0;
            }
        }
    }
}

// ---------------------------------------------------------------------------
// Transpose + f32->bf16 convert: dst[N][K] = (bf16) src[K][N]
// ---------------------------------------------------------------------------
__global__ __launch_bounds__(256) void transposeconv_kernel(
    const float* __restrict__ src, bf16* __restrict__ dst, int K, int N)
{
    __shared__ float tile[32][33];
    const int k0 = blockIdx.y * 32;
    const int n0 = blockIdx.x * 32;
    const int c  = threadIdx.x & 31;
    const int r  = threadIdx.x >> 5;   // 0..7
#pragma unroll
    for (int rr = 0; rr < 32; rr += 8)
        tile[r + rr][c] = src[(size_t)(k0 + r + rr) * N + n0 + c];
    __syncthreads();
#pragma unroll
    for (int rr = 0; rr < 32; rr += 8)
        dst[(size_t)(n0 + r + rr) * K + k0 + c] = (bf16)tile[c][r + rr];
}

// ---------------------------------------------------------------------------
// Token shift: dx = shifted - x ; tmix = bf16(x + dx*time_maa_x)
// ---------------------------------------------------------------------------
__global__ void timemix_kernel(const float* __restrict__ hidden,
                               const float* __restrict__ attn_x,
                               const float* __restrict__ maa_x,
                               float* __restrict__ dx, bf16* __restrict__ tmix)
{
    const size_t idx = (size_t)blockIdx.x * 256 + threadIdx.x;
    const int    c   = (int)(idx & (HID - 1));
    const size_t bt  = idx >> 11;            // / HID
    const int    t   = (int)(bt & (TT - 1));
    const int    b   = (int)(bt >> 11);
    const float  x    = hidden[idx];
    const float  prev = (t == 0) ? attn_x[b * HID + c] : hidden[idx - HID];
    const float  d    = prev - x;
    dx[idx]   = d;
    tmix[idx] = (bf16)(x + d * maa_x[c]);
}

// ---------------------------------------------------------------------------
// Decay path: only t = T-1 matters for new_attn_kv.
// wlast[b][c] = exp(-exp(time_decay[c] + tanh(xw_last @ w1) @ w2))
// ---------------------------------------------------------------------------
__global__ __launch_bounds__(256) void decay_kernel(
    const bf16* __restrict__ xw, const float* __restrict__ time_decay,
    const float* __restrict__ w1, const float* __restrict__ w2,
    float* __restrict__ wlast)
{
    __shared__ float h1[64];
    const int b = blockIdx.x;
    const bf16* xr = xw + (size_t)(b * TT + TT - 1) * HID;
    if (threadIdx.x < 64) {
        float s = 0.f;
        for (int c = 0; c < HID; ++c) s += (float)xr[c] * w1[c * 64 + threadIdx.x];
        h1[threadIdx.x] = tanhf(s);
    }
    __syncthreads();
    for (int c = threadIdx.x; c < HID; c += 256) {
        float s = time_decay[c];
        for (int j = 0; j < 64; ++j) s += h1[j] * w2[j * HID + c];
        wlast[b * HID + c] = __expf(-__expf(s));
    }
}

// ---------------------------------------------------------------------------
// Attention core per (b,h):
//   rk[t]     = sum_d r*faaaa*k
//   out[t][e] = rk[t]*v[t][e] + (r @ attn_kv)[t][e]   (WMMA on 2048x64 @ 64x64)
// attn_kv staged transposed (e-major) in LDS as bf16 so B frags are contiguous.
// ---------------------------------------------------------------------------
__global__ __launch_bounds__(256) void attention_kernel(
    const bf16* __restrict__ r, const bf16* __restrict__ k,
    const bf16* __restrict__ v, const float* __restrict__ attn_kv,
    const float* __restrict__ faaaa, float* __restrict__ out)
{
    __shared__ __align__(16) bf16 kvT[HEADD * HEADD];   // [e][d]
    __shared__ float fa[HEADD];
    __shared__ float rkbuf[8 * 16];

    const int b = blockIdx.x / NHEAD;
    const int h = blockIdx.x % NHEAD;
    const float* kv = attn_kv + (size_t)(b * NHEAD + h) * HEADD * HEADD;

    for (int i = threadIdx.x; i < HEADD * HEADD; i += 256) {
        const int d = i / HEADD, e = i % HEADD;
        kvT[e * HEADD + d] = (bf16)kv[i];
    }
    if (threadIdx.x < HEADD) fa[threadIdx.x] = faaaa[h * HEADD + threadIdx.x];
    __syncthreads();

    const int lane = threadIdx.x & 31;
    const int wave = threadIdx.x >> 5;
    const int half = lane >> 4;
    const int cl   = lane & 15;
    const size_t base_bh = (size_t)b * TT * HID + h * HEADD;

    for (int it = 0; it < TT / (16 * 8); ++it) {
        const int t0 = (it * 8 + wave) * 16;
        {   // rk for the 16 rows of this tile (lanes 0..15 publish)
            const int tr = t0 + cl;
            const bf16* rp = r + base_bh + (size_t)tr * HID;
            const bf16* kp = k + base_bh + (size_t)tr * HID;
            float s = 0.f;
            for (int d = 0; d < HEADD; ++d)
                s += (float)rp[d] * fa[d] * (float)kp[d];
            if (half == 0) rkbuf[wave * 16 + cl] = s;
        }
        __syncthreads();

        v8f acc[4] = {};
        for (int kk = 0; kk < HEADD; kk += 32) {
            FragB16 a;
            load_fragA(a, r + base_bh, HID, t0, kk, lane);
#pragma unroll
            for (int c = 0; c < 4; ++c) {
                FragB16 bb;
                const bf16* p = kvT + (c * 16 + cl) * HEADD + kk + half * 16;
                bb.q[0] = *(const uint4*)(p);
                bb.q[1] = *(const uint4*)(p + 8);
                acc[c] = __builtin_amdgcn_wmma_f32_16x16x32_bf16(
                    false, a.v, false, bb.v, (short)0, acc[c], false, false);
            }
        }
#pragma unroll
        for (int c = 0; c < 4; ++c) {
#pragma unroll
            for (int i = 0; i < 8; ++i) {
                const int tloc = i + 8 * half;
                const int t    = t0 + tloc;
                const int e    = c * 16 + cl;
                const size_t idx = base_bh + (size_t)t * HID + e;
                out[idx] = rkbuf[wave * 16 + tloc] * (float)v[idx] + acc[c][i];
            }
        }
        __syncthreads();
    }
}

// ---------------------------------------------------------------------------
// GroupNorm over (T, HEAD) per (b,h); fuse *g (silu gate) and emit bf16
// activation for the final output GEMM.
// ---------------------------------------------------------------------------
__global__ __launch_bounds__(256) void groupnorm_kernel(
    const float* __restrict__ out_pre, const bf16* __restrict__ g,
    const float* __restrict__ gamma, const float* __restrict__ beta,
    bf16* __restrict__ og)
{
    __shared__ float s1[256], s2[256];
    const int b = blockIdx.x / NHEAD;
    const int h = blockIdx.x % NHEAD;
    const size_t base = (size_t)b * TT * HID + h * HEADD;

    float sum = 0.f, sq = 0.f;
    for (int i = threadIdx.x; i < TT * HEADD; i += 256) {
        const int t = i / HEADD, e = i % HEADD;
        const float x = out_pre[base + (size_t)t * HID + e];
        sum += x; sq += x * x;
    }
    s1[threadIdx.x] = sum; s2[threadIdx.x] = sq;
    __syncthreads();
    for (int s = 128; s > 0; s >>= 1) {
        if (threadIdx.x < s) {
            s1[threadIdx.x] += s1[threadIdx.x + s];
            s2[threadIdx.x] += s2[threadIdx.x + s];
        }
        __syncthreads();
    }
    const float n    = (float)(TT * HEADD);
    const float mean = s1[0] / n;
    const float var  = s2[0] / n - mean * mean;
    const float rstd = rsqrtf(var + 1e-5f * (float)(HEADD * HEADD));

    for (int i = threadIdx.x; i < TT * HEADD; i += 256) {
        const int t = i / HEADD, e = i % HEADD;
        const int c = h * HEADD + e;
        const size_t idx = base + (size_t)t * HID + e;
        const float y = (out_pre[idx] - mean) * rstd * gamma[c] + beta[c];
        og[idx] = (bf16)(y * (float)g[idx]);
    }
}

// ---------------------------------------------------------------------------
// new_attn_kv[b,h,d,e] = k_last[d]*v_last[e] + w_last[d]*attn_kv[b,h,d,e]
// ---------------------------------------------------------------------------
__global__ void newkv_kernel(const bf16* __restrict__ k, const bf16* __restrict__ v,
                             const float* __restrict__ wlast,
                             const float* __restrict__ attn_kv,
                             float* __restrict__ dst)
{
    const int idx = blockIdx.x * 256 + threadIdx.x;     // B*NH*64*64 = 524288
    const int e = idx & 63;
    const int d = (idx >> 6) & 63;
    const int h = (idx >> 12) & 31;
    const int b = idx >> 17;
    const size_t lastrow = (size_t)(b * TT + TT - 1) * HID + h * HEADD;
    const float kl = (float)k[lastrow + d];
    const float vl = (float)v[lastrow + e];
    dst[idx] = kl * vl + wlast[b * HID + h * HEADD + d] * attn_kv[idx];
}

__global__ void lastrow_kernel(const float* __restrict__ hidden, float* __restrict__ dst)
{
    const int idx = blockIdx.x * 256 + threadIdx.x;     // 8192
    const int b = idx >> 11;
    const int c = idx & (HID - 1);
    dst[idx] = hidden[(size_t)(b * TT + TT - 1) * HID + c];
}

// ---------------------------------------------------------------------------
extern "C" void kernel_launch(void* const* d_in, const int* in_sizes, int n_in,
                              void* d_out, int out_size, void* d_ws, size_t ws_size,
                              hipStream_t stream)
{
    (void)in_sizes; (void)n_in; (void)out_size; (void)ws_size;

    const float* hidden   = (const float*)d_in[0];
    const float* attn_x   = (const float*)d_in[1];
    const float* attn_kv  = (const float*)d_in[2];
    const float* ffn_x    = (const float*)d_in[3];
    const float* maa_x    = (const float*)d_in[4];
    const float* maa_tab[5] = { (const float*)d_in[5], (const float*)d_in[6],
                                (const float*)d_in[7], (const float*)d_in[8],
                                (const float*)d_in[9] };           // w,k,v,r,g
    const float* maa_w1   = (const float*)d_in[10];                // [2048][160]
    const float* maa_w2   = (const float*)d_in[11];                // [5][32][2048]
    const float* tdecay   = (const float*)d_in[12];
    const float* tdec_w1  = (const float*)d_in[13];                // [2048][64]
    const float* tdec_w2  = (const float*)d_in[14];                // [64][2048]
    const float* faaaa    = (const float*)d_in[15];
    const float* bigW[5]  = { (const float*)d_in[16], (const float*)d_in[17],
                              (const float*)d_in[18], (const float*)d_in[19],
                              (const float*)d_in[20] };            // r,k,v,g,out
    const float* gamma    = (const float*)d_in[21];
    const float* beta     = (const float*)d_in[22];

    // ---- workspace layout (256B aligned) ----
    char* wp = (char*)d_ws;
    auto take = [&](size_t bytes) -> void* {
        void* p = wp;
        wp += (bytes + 255) & ~(size_t)255;
        return p;
    };
    float* dx    = (float*)take((size_t)MTOT * HID * 4);
    bf16*  tmix  = (bf16*) take((size_t)MTOT * HID * 2);
    bf16*  xxx   = (bf16*) take((size_t)MTOT * 160 * 2);
    bf16*  x5[5];
    for (int i = 0; i < 5; ++i) x5[i] = (bf16*)take((size_t)MTOT * HID * 2);
    bf16*  rb    = (bf16*) take((size_t)MTOT * HID * 2);
    bf16*  kb    = (bf16*) take((size_t)MTOT * HID * 2);
    bf16*  vb    = (bf16*) take((size_t)MTOT * HID * 2);
    bf16*  gb    = (bf16*) take((size_t)MTOT * HID * 2);
    bf16*  og    = (bf16*) take((size_t)MTOT * HID * 2);
    float* wlast = (float*)take((size_t)BB * HID * 4);
    bf16*  bigWt[5];
    for (int i = 0; i < 5; ++i) bigWt[i] = (bf16*)take((size_t)HID * HID * 2);
    bf16*  w1t   = (bf16*) take((size_t)160 * HID * 2);
    bf16*  w2t   = (bf16*) take((size_t)5 * HID * 32 * 2);

    // ---- d_out segments: out | new_attn_x | new_attn_kv | ffn_x ----
    float* out0    = (float*)d_out;                 // B*T*HID (also scratch pre-GN)
    float* out_ax  = out0 + (size_t)MTOT * HID;     // B*HID
    float* out_kv  = out_ax + BB * HID;             // B*NH*64*64
    float* out_ffn = out_kv + (size_t)BB * NHEAD * HEADD * HEADD;

    // 1) weight transpose + bf16 convert
    for (int i = 0; i < 5; ++i)
        transposeconv_kernel<<<dim3(HID / 32, HID / 32), 256, 0, stream>>>(
            bigW[i], bigWt[i], HID, HID);
    transposeconv_kernel<<<dim3(160 / 32, HID / 32), 256, 0, stream>>>(
        maa_w1, w1t, HID, 160);
    for (int n = 0; n < 5; ++n)
        transposeconv_kernel<<<dim3(HID / 32, 1), 256, 0, stream>>>(
            maa_w2 + (size_t)n * 32 * HID, w2t + (size_t)n * HID * 32, 32, HID);

    // 2) token shift
    timemix_kernel<<<(MTOT * HID) / 256, 256, 0, stream>>>(hidden, attn_x, maa_x, dx, tmix);

    // 3) xxx = tanh(tmix @ maa_w1)           (8192x160, K=2048), NT=2
    gemm_bf16<1, 2><<<dim3(160 / 32, MTOT / 128), 256, 0, stream>>>(
        tmix, HID, w1t, 160, HID, xxx, 160, nullptr, nullptr, nullptr, nullptr);

    // 4) x5[n] = x + dx*(maa_n + xxx_n @ w2_n)   (K=32), NT=4
    for (int n = 0; n < 5; ++n)
        gemm_bf16<3, 4><<<dim3(HID / 64, MTOT / 128), 256, 0, stream>>>(
            xxx + n * 32, 160, w2t + (size_t)n * HID * 32, HID, 32,
            x5[n], HID, nullptr, hidden, dx, maa_tab[n]);

    // 5) projections (xw=x5[0], xk=x5[1], xv=x5[2], xr=x5[3], xg=x5[4]), NT=4
    gemm_bf16<0, 4><<<dim3(HID / 64, MTOT / 128), 256, 0, stream>>>(
        x5[3], HID, bigWt[0], HID, HID, rb, HID, nullptr, nullptr, nullptr, nullptr);
    gemm_bf16<0, 4><<<dim3(HID / 64, MTOT / 128), 256, 0, stream>>>(
        x5[1], HID, bigWt[1], HID, HID, kb, HID, nullptr, nullptr, nullptr, nullptr);
    gemm_bf16<0, 4><<<dim3(HID / 64, MTOT / 128), 256, 0, stream>>>(
        x5[2], HID, bigWt[2], HID, HID, vb, HID, nullptr, nullptr, nullptr, nullptr);
    gemm_bf16<2, 4><<<dim3(HID / 64, MTOT / 128), 256, 0, stream>>>(
        x5[4], HID, bigWt[3], HID, HID, gb, HID, nullptr, nullptr, nullptr, nullptr);

    // 6) decay (last timestep only)
    decay_kernel<<<BB, 256, 0, stream>>>(x5[0], tdecay, tdec_w1, tdec_w2, wlast);

    // 7) attention core -> out0 (pre-groupnorm scratch)
    attention_kernel<<<BB * NHEAD, 256, 0, stream>>>(rb, kb, vb, attn_kv, faaaa, out0);

    // 8) groupnorm + gate fuse -> og (bf16)
    groupnorm_kernel<<<BB * NHEAD, 256, 0, stream>>>(out0, gb, gamma, beta, og);

    // 9) final projection: out = og @ output_w  (overwrites out0)
    gemm_bf16<4, 4><<<dim3(HID / 64, MTOT / 128), 256, 0, stream>>>(
        og, HID, bigWt[4], HID, HID, nullptr, HID, out0, nullptr, nullptr, nullptr);

    // 10) states
    newkv_kernel<<<(BB * NHEAD * HEADD * HEADD) / 256, 256, 0, stream>>>(
        kb, vb, wlast, attn_kv, out_kv);
    lastrow_kernel<<<(BB * HID) / 256, 256, 0, stream>>>(hidden, out_ax);
    hipMemcpyAsync(out_ffn, ffn_x, (size_t)BB * HID * sizeof(float),
                   hipMemcpyDeviceToDevice, stream);
}
